// Block_80736795230695
// MI455X (gfx1250) — matrix operand order repaired and checked by reading, not compiled, not run on hardware
//
#include <hip/hip_runtime.h>
#include <hip/hip_bf16.h>
#include <math.h>

// ---------------- problem constants (fixed by the harness) ----------------
#define BATCH 16
#define HH    32
#define WW    32
#define NPOS  (BATCH*HH*WW)   // 16384 rows
#define DIMC  384
#define HID   1536
#define NB    8
#define BS    48
#define WF    17               // W/2 + 1

typedef __attribute__((ext_vector_type(16))) _Float16 v16h;
typedef __attribute__((ext_vector_type(8)))  _Float16 v8h;
typedef __attribute__((ext_vector_type(8)))  float    v8f;

// CDNA5 async global->LDS path (ASYNCcnt-tracked). Guarded: falls back to a
// synchronous LDS staging copy if the toolchain lacks the builtins.
// Probe result (round 2 diagnostic): signature is
//   void __builtin_amdgcn_global_load_async_to_lds_b128(int4 __device__* src,
//                                                       int4 __shared__* dst,
//                                                       int imm_offset, int cpol);
#ifndef __has_builtin
#define __has_builtin(x) 0
#endif
#if __has_builtin(__builtin_amdgcn_global_load_async_to_lds_b128) && \
    __has_builtin(__builtin_amdgcn_s_wait_asynccnt)
#define USE_ASYNC_LDS 1
typedef int v4i __attribute__((ext_vector_type(4)));
typedef __attribute__((address_space(1))) v4i as1_v4i;
typedef __attribute__((address_space(3))) v4i as3_v4i;
#else
#define USE_ASYNC_LDS 0
#endif

// ---------------- layernorm (wave32: one wave per 384-wide row) ----------------
__global__ __launch_bounds__(256) void ln_f32_kernel(const float* __restrict__ in,
                                                     const float* __restrict__ g,
                                                     const float* __restrict__ b,
                                                     float* __restrict__ out) {
  int lane = threadIdx.x & 31;
  int row  = blockIdx.x * 8 + (threadIdx.x >> 5);
  const float* r = in + (size_t)row * DIMC;
  float vals[12];
  float s = 0.f, s2 = 0.f;
#pragma unroll
  for (int i = 0; i < 12; ++i) { float v = r[lane + 32*i]; vals[i] = v; s += v; s2 += v*v; }
#pragma unroll
  for (int off = 16; off > 0; off >>= 1) { s += __shfl_xor(s, off, 32); s2 += __shfl_xor(s2, off, 32); }
  float mean = s * (1.0f/DIMC);
  float var  = s2 * (1.0f/DIMC) - mean*mean;
  float inv  = rsqrtf(var + 1e-5f);
  float* o = out + (size_t)row * DIMC;
#pragma unroll
  for (int i = 0; i < 12; ++i) { int c = lane + 32*i; o[c] = (vals[i]-mean)*inv*g[c] + b[c]; }
}

__global__ __launch_bounds__(256) void ln_f16_kernel(const float* __restrict__ in,
                                                     const float* __restrict__ g,
                                                     const float* __restrict__ b,
                                                     _Float16* __restrict__ out) {
  int lane = threadIdx.x & 31;
  int row  = blockIdx.x * 8 + (threadIdx.x >> 5);
  const float* r = in + (size_t)row * DIMC;
  float vals[12];
  float s = 0.f, s2 = 0.f;
#pragma unroll
  for (int i = 0; i < 12; ++i) { float v = r[lane + 32*i]; vals[i] = v; s += v; s2 += v*v; }
#pragma unroll
  for (int off = 16; off > 0; off >>= 1) { s += __shfl_xor(s, off, 32); s2 += __shfl_xor(s2, off, 32); }
  float mean = s * (1.0f/DIMC);
  float var  = s2 * (1.0f/DIMC) - mean*mean;
  float inv  = rsqrtf(var + 1e-5f);
  _Float16* o = out + (size_t)row * DIMC;
#pragma unroll
  for (int i = 0; i < 12; ++i) { int c = lane + 32*i; o[c] = (_Float16)((vals[i]-mean)*inv*g[c] + b[c]); }
}

// ---------------- FFT helpers: exact 32-point DFTs with LDS twiddles ----------------
__global__ __launch_bounds__(256) void rfft_w_kernel(const float* __restrict__ n1,
                                                     float* __restrict__ outr,
                                                     float* __restrict__ outi) {
  __shared__ float tc[32], ts[32];
  if (threadIdx.x < 32) {
    float a = 0.19634954084936207f * (float)threadIdx.x;  // 2*pi/32
    tc[threadIdx.x] = cosf(a); ts[threadIdx.x] = sinf(a);
  }
  __syncthreads();
  int tid = blockIdx.x * blockDim.x + threadIdx.x;         // (b,h,kw,c) row-major
  int c = tid % DIMC; int t = tid / DIMC;
  int kw = t % WF; t /= WF;
  int h = t % HH; int b = t / HH;
  const float* base = n1 + ((size_t)(b*HH + h) * WW) * DIMC + c;
  float ar = 0.f, ai = 0.f;
  for (int w = 0; w < WW; ++w) {
    float v = base[(size_t)w * DIMC];
    int idx = (w * kw) & 31;
    ar += v * tc[idx];
    ai -= v * ts[idx];
  }
  outr[tid] = ar; outi[tid] = ai;
}

__global__ __launch_bounds__(256) void fft_h_kernel(const float* __restrict__ inr,
                                                    const float* __restrict__ ini,
                                                    float* __restrict__ outr,
                                                    float* __restrict__ outi) {
  __shared__ float tc[32], ts[32];
  if (threadIdx.x < 32) {
    float a = 0.19634954084936207f * (float)threadIdx.x;
    tc[threadIdx.x] = cosf(a); ts[threadIdx.x] = sinf(a);
  }
  __syncthreads();
  int tid = blockIdx.x * blockDim.x + threadIdx.x;         // (b,kh,kw,c)
  int c = tid % DIMC; int t = tid / DIMC;
  int kw = t % WF; t /= WF;
  int kh = t % HH; int b = t / HH;
  const size_t stride = (size_t)WF * DIMC;
  const float* br = inr + (size_t)b*HH*stride + (size_t)kw*DIMC + c;
  const float* bi = ini + (size_t)b*HH*stride + (size_t)kw*DIMC + c;
  float ar = 0.f, ai = 0.f;
  for (int h = 0; h < HH; ++h) {
    float xr = br[(size_t)h*stride], xi = bi[(size_t)h*stride];
    int idx = (h * kh) & 31;
    float cc = tc[idx], ss = ts[idx];            // e^{-i t} = (cc, -ss)
    ar += xr*cc + xi*ss;
    ai += xi*cc - xr*ss;
  }
  outr[tid] = ar * 0.03125f; outi[tid] = ai * 0.03125f;    // full forward ortho scale
}

__global__ __launch_bounds__(256) void ifft_h_kernel(const float* __restrict__ inr,
                                                     const float* __restrict__ ini,
                                                     float* __restrict__ outr,
                                                     float* __restrict__ outi) {
  __shared__ float tc[32], ts[32];
  if (threadIdx.x < 32) {
    float a = 0.19634954084936207f * (float)threadIdx.x;
    tc[threadIdx.x] = cosf(a); ts[threadIdx.x] = sinf(a);
  }
  __syncthreads();
  int tid = blockIdx.x * blockDim.x + threadIdx.x;         // (b,h,kw,c)
  int c = tid % DIMC; int t = tid / DIMC;
  int kw = t % WF; t /= WF;
  int h = t % HH; int b = t / HH;
  const size_t stride = (size_t)WF * DIMC;
  const float* br = inr + (size_t)b*HH*stride + (size_t)kw*DIMC + c;
  const float* bi = ini + (size_t)b*HH*stride + (size_t)kw*DIMC + c;
  float ar = 0.f, ai = 0.f;
  for (int kh = 0; kh < HH; ++kh) {
    float xr = br[(size_t)kh*stride], xi = bi[(size_t)kh*stride];
    int idx = (h * kh) & 31;
    float cc = tc[idx], ss = ts[idx];            // e^{+i t} = (cc, +ss)
    ar += xr*cc - xi*ss;
    ai += xi*cc + xr*ss;
  }
  outr[tid] = ar; outi[tid] = ai;
}

__global__ __launch_bounds__(256) void irfft_w_residual_kernel(const float* __restrict__ inr,
                                                               const float* __restrict__ ini,
                                                               const float* __restrict__ x,
                                                               const float* __restrict__ n1,
                                                               float* __restrict__ x1) {
  __shared__ float tc[32], ts[32];
  if (threadIdx.x < 32) {
    float a = 0.19634954084936207f * (float)threadIdx.x;
    tc[threadIdx.x] = cosf(a); ts[threadIdx.x] = sinf(a);
  }
  __syncthreads();
  int tid = blockIdx.x * blockDim.x + threadIdx.x;         // (b,h,w,c) == index into x
  int c = tid % DIMC; int t = tid / DIMC;
  int w = t % WW; t /= WW;
  int h = t % HH; int b = t / HH;
  size_t base = ((size_t)(b*HH + h) * WF) * DIMC + c;
  float acc = 0.f;
  for (int kw = 0; kw < WF; ++kw) {
    float alpha = (kw == 0 || kw == 16) ? 1.f : 2.f;
    int idx = (w * kw) & 31;
    float xr = inr[base + (size_t)kw*DIMC];
    float xi = ini[base + (size_t)kw*DIMC];
    acc += alpha * (xr * tc[idx] - xi * ts[idx]);  // Re[(xr+ixi) e^{+i t}]
  }
  acc *= 0.03125f;                                  // ortho inverse scale 1/32
  x1[tid] = x[tid] + n1[tid] + acc;
}

// ---------------- AFNO frequency-domain block MLP + softshrink ----------------
__global__ __launch_bounds__(384) void afno_mlp_kernel(const float* __restrict__ inr,
                                                       const float* __restrict__ ini,
                                                       float* __restrict__ outr,
                                                       float* __restrict__ outi,
                                                       const float* __restrict__ w1,
                                                       const float* __restrict__ b1,
                                                       const float* __restrict__ w2,
                                                       const float* __restrict__ b2) {
  __shared__ float xr[DIMC], xi[DIMC], o1r[DIMC], o1i[DIMC];
  int t = threadIdx.x;
  size_t p = (size_t)blockIdx.x * DIMC;
  xr[t] = inr[p + t]; xi[t] = ini[p + t];
  __syncthreads();
  int n = t / BS, o = t % BS;
  // layer 1: w1[ri][n][i][o]
  const float* w1r = w1 + (size_t)(n*BS)*BS + o;
  const float* w1i = w1 + (size_t)((NB + n)*BS)*BS + o;
  float accr = b1[n*BS + o];
  float acci = b1[(NB + n)*BS + o];
  const float* xrn = xr + n*BS; const float* xin = xi + n*BS;
  for (int i = 0; i < BS; ++i) {
    float wr = w1r[i*BS], wi = w1i[i*BS];
    float vr = xrn[i],    vi = xin[i];
    accr += vr*wr - vi*wi;
    acci += vi*wr + vr*wi;
  }
  o1r[t] = fmaxf(accr, 0.f);
  o1i[t] = fmaxf(acci, 0.f);
  __syncthreads();
  // layer 2: w2[ri][n][o][i] ; this thread produces i = o
  const float* w2r = w2 + (size_t)(n*BS)*BS + o;
  const float* w2i = w2 + (size_t)((NB + n)*BS)*BS + o;
  float a2r = b2[n*BS + o];
  float a2i = b2[(NB + n)*BS + o];
  const float* orn = o1r + n*BS; const float* oin = o1i + n*BS;
  for (int oo = 0; oo < BS; ++oo) {
    float wr = w2r[oo*BS], wi = w2i[oo*BS];
    float vr = orn[oo],    vi = oin[oo];
    a2r += vr*wr - vi*wi;
    a2i += vi*wr + vr*wi;
  }
  // softshrink(lambda = 0.01)
  float rr = a2r > 0.01f ? a2r - 0.01f : (a2r < -0.01f ? a2r + 0.01f : 0.f);
  float ii = a2i > 0.01f ? a2i - 0.01f : (a2i < -0.01f ? a2i + 0.01f : 0.f);
  outr[p + t] = rr;
  outi[p + t] = ii;
}

// ---------------- misc elementwise ----------------
__global__ __launch_bounds__(256) void f32_to_f16_kernel(const float* __restrict__ src,
                                                         _Float16* __restrict__ dst, int n) {
  int i = blockIdx.x * blockDim.x + threadIdx.x;
  if (i < n) dst[i] = (_Float16)src[i];
}

// depthwise 3x3 (f16 input, f32 accumulate) + bias + exact GELU -> f16
__global__ __launch_bounds__(256) void dwconv_gelu_kernel(const _Float16* __restrict__ h,
                                                          const float* __restrict__ wgt,
                                                          const float* __restrict__ bias,
                                                          _Float16* __restrict__ out) {
  size_t tid = (size_t)blockIdx.x * blockDim.x + threadIdx.x;  // (b,hh,ww,ch)
  int ch = (int)(tid % HID); size_t pos = tid / HID;
  int w  = (int)(pos % WW); int hh = (int)((pos / WW) % HH); int b = (int)(pos / (WW*HH));
  float acc = bias[ch];
  const float* wp = wgt + (size_t)ch * 9;
  for (int dh = 0; dh < 3; ++dh) {
    int yy = hh + dh - 1;
    if (yy < 0 || yy >= HH) continue;
    for (int dw = 0; dw < 3; ++dw) {
      int xx = w + dw - 1;
      if (xx < 0 || xx >= WW) continue;
      acc += (float)h[(((size_t)b*HH + yy)*WW + xx)*HID + ch] * wp[dh*3 + dw];
    }
  }
  float g = 0.5f * acc * (1.0f + erff(acc * 0.7071067811865476f)); // exact gelu
  out[tid] = (_Float16)g;
}

// ---------------- WMMA GEMM with LDS-staged B tiles ------------------------------
// out[M,N] = A[M,K](f16) * B[K,N](f16) + bias (+resid). Wave tile 16x64 (4 acc),
// block = 8 waves = 128x64. The 32x64 B tile is staged once per block through LDS
// (8x less global B traffic); CDNA5 async global->LDS copy when available.
// A fragment per lane: two contiguous 16B global loads (ISA 7.12.2 16-bit A 16x32).
// B fragment per lane: column lane&15 of the LDS tile, 16 strided 2-byte reads (rows 8h+i, 16+8h+i).
__global__ __launch_bounds__(256) void wmma_gemm_kernel(const _Float16* __restrict__ A,
                                                        const _Float16* __restrict__ Bw,
                                                        const float* __restrict__ bias,
                                                        const float* __restrict__ resid,
                                                        float* __restrict__ outf,
                                                        _Float16* __restrict__ outh,
                                                        int K, int N) {
  __shared__ _Float16 sB[32 * 64];                 // 4 KB tile
  int t    = threadIdx.x;
  int lane = t & 31;
  int wav  = t >> 5;
  int m0 = blockIdx.y * 128 + wav * 16;
  int n0 = blockIdx.x * 64;
  v8f acc0 = {}, acc1 = {}, acc2 = {}, acc3 = {};
  const _Float16* ap = A + (size_t)(m0 + (lane & 15)) * K + ((lane >> 4) << 3);
  // staging assignment: thread t copies 16B: row = t>>3, col chunk = (t&7)*8 halves
  const int srow = t >> 3, scol = (t & 7) * 8;
  _Float16* ldst = sB + srow * 64 + scol;
  const _Float16* brow = sB + (lane & 15) + ((lane >> 4) << 3) * 64; // per-lane B fragment base

  for (int k0 = 0; k0 < K; k0 += 32) {
    const _Float16* gsrc = Bw + (size_t)(k0 + srow) * N + n0 + scol;
#if USE_ASYNC_LDS
    __builtin_amdgcn_global_load_async_to_lds_b128((as1_v4i*)gsrc, (as3_v4i*)ldst, 0, 0);
    __builtin_amdgcn_s_wait_asynccnt(0);
#else
    *(v8h*)ldst = *(const v8h*)gsrc;
#endif
    __syncthreads();

    v8h alo = *(const v8h*)(ap + k0);        // K {0..7} or {8..15}
    v8h ahi = *(const v8h*)(ap + k0 + 16);   // K {16..23} or {24..31}
    v16h a;
#pragma unroll
    for (int i = 0; i < 8; ++i) { a[i] = alo[i]; a[i+8] = ahi[i]; }

    v16h bf0, bf1, bf2, bf3;
#pragma unroll
    for (int i = 0; i < 8; ++i) {
      const _Float16* p0 = brow + i * 64;
      const _Float16* p1 = brow + (16 + i) * 64;
      bf0[i] = p0[0];  bf0[i+8] = p1[0];
      bf1[i] = p0[16]; bf1[i+8] = p1[16];
      bf2[i] = p0[32]; bf2[i+8] = p1[32];
      bf3[i] = p0[48]; bf3[i+8] = p1[48];
    }
    acc0 = __builtin_amdgcn_wmma_f32_16x16x32_f16(false, a, false, bf0, (short)0, acc0, false, false);
    acc1 = __builtin_amdgcn_wmma_f32_16x16x32_f16(false, a, false, bf1, (short)0, acc1, false, false);
    acc2 = __builtin_amdgcn_wmma_f32_16x16x32_f16(false, a, false, bf2, (short)0, acc2, false, false);
    acc3 = __builtin_amdgcn_wmma_f32_16x16x32_f16(false, a, false, bf3, (short)0, acc3, false, false);
    __syncthreads();                          // before next tile overwrite
  }

  // D layout: VGPR r -> M = m0 + r (lanes 0-15) / m0 + 8 + r (lanes 16-31); N = n0 + (lane&15)
  int mr = m0 + ((lane >> 4) << 3);
  int nc = n0 + (lane & 15);
  v8f accs[4] = {acc0, acc1, acc2, acc3};
#pragma unroll
  for (int j = 0; j < 4; ++j) {
    int c = nc + 16*j;
    float bv = bias[c];
#pragma unroll
    for (int r = 0; r < 8; ++r) {
      size_t idx = (size_t)(mr + r) * N + c;
      float v = accs[j][r] + bv;
      if (outh) {
        outh[idx] = (_Float16)v;
      } else {
        if (resid) v += resid[idx];
        outf[idx] = v;
      }
    }
  }
}

// ---------------- launcher ----------------
extern "C" void kernel_launch(void* const* d_in, const int* in_sizes, int n_in,
                              void* d_out, int out_size, void* d_ws, size_t ws_size,
                              hipStream_t stream) {
  (void)in_sizes; (void)n_in; (void)out_size; (void)ws_size;
  const float* x      = (const float*)d_in[0];
  const float* ln1_g  = (const float*)d_in[3];
  const float* ln1_b  = (const float*)d_in[4];
  const float* ln2_g  = (const float*)d_in[5];
  const float* ln2_b  = (const float*)d_in[6];
  const float* afno_w1= (const float*)d_in[7];
  const float* afno_b1= (const float*)d_in[8];
  const float* afno_w2= (const float*)d_in[9];
  const float* afno_b2= (const float*)d_in[10];
  const float* fc1_w  = (const float*)d_in[11];
  const float* fc1_b  = (const float*)d_in[12];
  const float* dw_w   = (const float*)d_in[13];
  const float* dw_b   = (const float*)d_in[14];
  const float* fc2_w  = (const float*)d_in[15];
  const float* fc2_b  = (const float*)d_in[16];
  float* out = (float*)d_out;

  // workspace layout (bytes, all 256-aligned); total ~157 MB
  char* ws = (char*)d_ws;
  const size_t XBYTES = (size_t)NPOS * DIMC * 4;              // 25165824
  const size_t FREQB  = (size_t)BATCH * HH * WF * DIMC * 4;   // 13369344
  const size_t HB16   = (size_t)NPOS * HID * 2;               // 50331648
  float*    n1   = (float*)(ws);
  float*    x1   = (float*)(ws + XBYTES);
  float*    Ar   = (float*)(ws + 2*XBYTES);
  float*    Ai   = (float*)(ws + 2*XBYTES + FREQB);
  float*    Br   = (float*)(ws + 2*XBYTES + 2*FREQB);
  float*    Bi   = (float*)(ws + 2*XBYTES + 3*FREQB);
  _Float16* acth = (_Float16*)(ws + 2*XBYTES + 4*FREQB);      // post-gelu f16 [NPOS,HID]
  _Float16* w1h  = (_Float16*)(ws + 2*XBYTES + 4*FREQB + HB16);
  _Float16* w2h  = (_Float16*)(ws + 2*XBYTES + 4*FREQB + HB16 + (size_t)DIMC*HID*2);
  // reuse dead regions:
  _Float16* n2h  = (_Float16*)(ws);              // aliases n1 (dead after irfft fuse)
  _Float16* h16a = (_Float16*)(ws + 2*XBYTES);   // fc1 out f16; aliases freq bufs (dead after irfft)

  // 0) weight conversion to f16 (row-major [K][N], natural WMMA-B layout)
  f32_to_f16_kernel<<<(DIMC*HID + 255)/256, 256, 0, stream>>>(fc1_w, w1h, DIMC*HID);
  f32_to_f16_kernel<<<(HID*DIMC + 255)/256, 256, 0, stream>>>(fc2_w, w2h, HID*DIMC);

  // 1) n1 = LN1(x)
  ln_f32_kernel<<<NPOS/8, 256, 0, stream>>>(x, ln1_g, ln1_b, n1);

  // 2) AFNO forward FFTs (ortho; 1/32 applied in fft_h)
  const int FREQTH = BATCH*HH*WF*DIMC;          // 3342336
  rfft_w_kernel<<<FREQTH/256, 256, 0, stream>>>(n1, Ar, Ai);
  fft_h_kernel<<<FREQTH/256, 256, 0, stream>>>(Ar, Ai, Br, Bi);

  // 3) frequency-domain block MLP + softshrink
  afno_mlp_kernel<<<BATCH*HH*WF, 384, 0, stream>>>(Br, Bi, Ar, Ai,
                                                   afno_w1, afno_b1, afno_w2, afno_b2);

  // 4) inverse FFTs + fused residual: x1 = x + n1 + irfft2(...)
  ifft_h_kernel<<<FREQTH/256, 256, 0, stream>>>(Ar, Ai, Br, Bi);
  irfft_w_residual_kernel<<<(NPOS*DIMC)/256, 256, 0, stream>>>(Br, Bi, x, n1, x1);

  // 5) n2 = LN2(x1) -> f16 GEMM operand
  ln_f16_kernel<<<NPOS/8, 256, 0, stream>>>(x1, ln2_g, ln2_b, n2h);

  // 6) fc1: h16a(f16) = n2h @ w1h + fc1_b   (M=16384, K=384, N=1536)
  {
    dim3 g(HID/64, NPOS/128);
    wmma_gemm_kernel<<<g, 256, 0, stream>>>(n2h, w1h, fc1_b, nullptr, nullptr, h16a, DIMC, HID);
  }

  // 7) depthwise 3x3 conv + bias + exact GELU -> f16
  dwconv_gelu_kernel<<<(NPOS*HID)/256, 256, 0, stream>>>(h16a, dw_w, dw_b, acth);

  // 8) fc2 + residual: out = x1 + acth @ w2h + fc2_b   (M=16384, K=1536, N=384)
  {
    dim3 g(DIMC/64, NPOS/128);
    wmma_gemm_kernel<<<g, 256, 0, stream>>>(acth, w2h, fc2_b, x1, out, nullptr, HID, DIMC);
  }
}
